// DeepKnowledgeTracing_1554778161825
// MI455X (gfx1250) — compile-verified, matrix operand order checked by reading, not run on hardware
//
#include <hip/hip_runtime.h>
#include <hip/hip_bf16.h>
#include <math.h>

// Problem constants (match reference)
#define B_  32
#define T_  200
#define H_  2048
#define S_  256
#define K4  4096   // 2H : [x | h] concat depth
#define G4  8192   // 4H : gate width (permuted: n' = 4*j + gate)

typedef __attribute__((ext_vector_type(16))) __bf16 v16bf;
typedef __attribute__((ext_vector_type(8)))  __bf16 v8bf;
typedef __attribute__((ext_vector_type(8)))  float  v8f;

// K index inside a 32-deep WMMA fragment for element e (0..15) of a lane in
// half `half` (lane>>4), per CDNA5 ISA 7.12.2 16-bit A/B layouts.
__device__ __forceinline__ int kmap(int e, int half) {
    int v = e >> 1, p = e & 1;
    return ((v >> 2) << 4) + ((v & 3) << 1) + p + half * 8;
}

// Load a 16-bit A fragment: two contiguous 16B runs at base and base+32B.
__device__ __forceinline__ v16bf load_afrag(const __bf16* __restrict__ base) {
    v8bf lo = *(const v8bf*)base;
    v8bf hi = *(const v8bf*)(base + 16);
    v16bf a;
#pragma unroll
    for (int i = 0; i < 8; ++i) { a[i] = lo[i]; a[8 + i] = hi[i]; }
    return a;
}

// ---------------------------------------------------------------------------
// Pack fp32 weights into bf16 WMMA B-fragment tiles. tile(nt,kt) = 512
// contiguous bf16 laid out [lane][e] (one 32B read per lane per tile).
// perm_gates!=0 applies the i/f/g/o interleave: output column n' = 4*j + g
// maps to original output row g*H + j of [Wih|Whh].
// ---------------------------------------------------------------------------
__global__ void pack_b_kernel(const float* __restrict__ W0,
                              const float* __restrict__ W1,
                              int split, int K, int N, int perm_gates,
                              __bf16* __restrict__ Bp)
{
    long idx = (long)blockIdx.x * blockDim.x + threadIdx.x;
    long total = (long)K * N;
    if (idx >= total) return;
    int  e    = (int)(idx & 15);
    int  lane = (int)((idx >> 4) & 31);
    long tile = idx >> 9;
    int  ktiles = K / 32;
    int  kt = (int)(tile % ktiles);
    int  nt = (int)(tile / ktiles);
    int  n    = nt * 16 + (lane & 15);
    int  half = lane >> 4;
    int  k    = kt * 32 + kmap(e, half);
    int  row  = perm_gates ? ((n & 3) * H_ + (n >> 2)) : n;
    float val = (k < split) ? W0[(long)row * split + k]
                            : W1[(long)row * split + (k - split)];
    Bp[idx] = (__bf16)val;
}

// Permuted combined bias: pbias[n'] = bih[row] + bhh[row], row = (n'&3)*H + n'>>2.
__global__ void pbias_kernel(const float* __restrict__ bih,
                             const float* __restrict__ bhh,
                             float* __restrict__ pbias)
{
    int n = blockIdx.x * 256 + threadIdx.x;
    if (n >= G4) return;
    int row = (n & 3) * H_ + (n >> 2);
    pbias[n] = bih[row] + bhh[row];
}

// ---------------------------------------------------------------------------
// Precompute all encoder outputs X[t][b][h] (recurrence-independent).
// ---------------------------------------------------------------------------
__global__ void encoder_pre_kernel(const float* __restrict__ x1,
                                   const float* __restrict__ x2,
                                   const int*   __restrict__ skill,
                                   const float* __restrict__ m1W, const float* __restrict__ m1b,
                                   const float* __restrict__ m2W, const float* __restrict__ m2b,
                                   const float* __restrict__ encW, const float* __restrict__ encb,
                                   __bf16* __restrict__ Xpre)
{
    int tb = blockIdx.x;          // t*B_ + b
    int t = tb / B_, b = tb % B_;
    float a0 = x1[((long)b * T_ + t) * 2 + 0];
    float a1 = x1[((long)b * T_ + t) * 2 + 1];
    float a2 = x2[(long)b * T_ + t];
    float f[20];
#pragma unroll
    for (int i = 0; i < 10; ++i) f[i]      = m1W[i * 2] * a0 + m1W[i * 2 + 1] * a1 + m1b[i];
#pragma unroll
    for (int i = 0; i < 10; ++i) f[10 + i] = m2W[i] * a2 + m2b[i];

    int sk = skill[(long)b * T_ + t];
    const float* We = encW + (long)sk * 20 * H_;
    const float* be = encb + (long)sk * H_;
    for (int h = threadIdx.x; h < H_; h += blockDim.x) {
        float acc = be[h];
#pragma unroll
        for (int i = 0; i < 20; ++i) acc += f[i] * We[(long)i * H_ + h];
        Xpre[((long)t * B_ + b) * H_ + h] = (__bf16)acc;
    }
}

// Initialize xh[0] = [X[0] | bf16(h0)], c state, and barrier counters.
__global__ void init_kernel(const float* __restrict__ h0, const float* __restrict__ c0,
                            const __bf16* __restrict__ Xpre,
                            __bf16* __restrict__ xhbuf, float* __restrict__ cbuf,
                            unsigned* __restrict__ cnt, unsigned* __restrict__ sense)
{
    int idx = blockIdx.x * 256 + threadIdx.x;
    if (idx == 0) { *cnt = 0u; *sense = 0u; }
    if (idx >= B_ * H_) return;
    int b = idx / H_, j = idx % H_;
    xhbuf[(long)b * K4 + j]      = Xpre[idx];
    xhbuf[(long)b * K4 + H_ + j] = (__bf16)h0[idx];
    cbuf[idx] = c0[idx];
}

// ---------------------------------------------------------------------------
// Device-wide sense-reversal barrier (all blocks resident; grid is 66 blocks).
// ---------------------------------------------------------------------------
__device__ __forceinline__ void grid_sync(unsigned* cnt, unsigned* sense,
                                          unsigned phase, unsigned nblk)
{
    __threadfence();
    __syncthreads();
    if (threadIdx.x == 0) {
        unsigned old = __hip_atomic_fetch_add(cnt, 1u, __ATOMIC_ACQ_REL, __HIP_MEMORY_SCOPE_AGENT);
        if (old == nblk - 1u) {
            __hip_atomic_store(cnt, 0u, __ATOMIC_RELAXED, __HIP_MEMORY_SCOPE_AGENT);
            __hip_atomic_store(sense, phase, __ATOMIC_RELEASE, __HIP_MEMORY_SCOPE_AGENT);
        } else {
            while (__hip_atomic_load(sense, __ATOMIC_ACQUIRE, __HIP_MEMORY_SCOPE_AGENT) < phase) {
                __builtin_amdgcn_s_sleep(2);
            }
        }
    }
    __syncthreads();
}

// ---------------------------------------------------------------------------
// Persistent recurrence kernel: 528 waves (512 gate + 16 decoder), one grid
// barrier per step. Gate wave nt owns output columns n' in [16nt,16nt+16) =
// all four gates of hidden units j in [4nt,4nt+4): after the K-loop it
// finishes the LSTM cell in registers (quad shuffle + activations) and writes
// h_new into the ping-pong xh buffer. Decoder waves emit out[t-1] from the
// same resident h.
// ---------------------------------------------------------------------------
__global__ __launch_bounds__(256, 1)
void dkt_persistent_kernel(const __bf16* __restrict__ Blstm,
                           const __bf16* __restrict__ Bdec,
                           const float*  __restrict__ pbias,
                           const float*  __restrict__ decb,
                           const __bf16* __restrict__ Xpre,
                           __bf16* __restrict__ xhbuf,   // 2 * B_*K4 bf16
                           float*  __restrict__ cbuf,    // B_*H_ fp32
                           float*  __restrict__ out,     // d_out
                           unsigned* __restrict__ cnt,
                           unsigned* __restrict__ sense)
{
    const int wid  = blockIdx.x * 8 + (threadIdx.x >> 5);
    const int lane = threadIdx.x & 31;
    const int l16  = lane & 15;
    const int half = lane >> 4;
    const unsigned nblk = gridDim.x;

    const bool isGate = (wid < G4 / 16);
    const bool isDec  = !isGate && (wid < G4 / 16 + S_ / 16);
    const int  nt     = isGate ? wid : (wid - G4 / 16);

    const float biasv = isGate ? pbias[nt * 16 + l16] : 0.0f;
    const float dbias = isDec  ? decb[nt * 16 + l16]  : 0.0f;

    const int aGate = (lane >> 2) & 3;           // which j within the quad
    const int jGate = nt * 4 + aGate;            // hidden unit (gate waves)
    const bool storer = (lane & 3) == 0;

    for (int t = 0; t <= T_; ++t) {
        const __bf16* xh  = xhbuf + (size_t)(t & 1)       * (B_ * K4);
        __bf16*       xhn = xhbuf + (size_t)((t + 1) & 1) * (B_ * K4);

        if (isGate && t < T_) {
            // ---- gates GEMM: [32,4096] @ packed B column nt, K pipelined ----
            const __bf16* Bp = Blstm + (size_t)nt * (K4 / 32) * 512 + lane * 16;
            const __bf16* arow0 = xh + (size_t)l16 * K4 + half * 8;
            const __bf16* arow1 = arow0 + (size_t)16 * K4;
            v8f accs[2] = {v8f{}, v8f{}};
            v16bf bc  = *(const v16bf*)Bp;
            v16bf a0c = load_afrag(arow0);
            v16bf a1c = load_afrag(arow1);
            for (int kt = 0; kt < K4 / 32; ++kt) {
                v16bf bn = bc, a0n = a0c, a1n = a1c;
                if (kt + 1 < K4 / 32) {
                    bn  = *(const v16bf*)(Bp + (size_t)(kt + 1) * 512);
                    a0n = load_afrag(arow0 + (kt + 1) * 32);
                    a1n = load_afrag(arow1 + (kt + 1) * 32);
                }
                accs[0] = __builtin_amdgcn_wmma_f32_16x16x32_bf16(false, a0c, false, bc, (short)0, accs[0], false, false);
                accs[1] = __builtin_amdgcn_wmma_f32_16x16x32_bf16(false, a1c, false, bc, (short)0, accs[1], false, false);
                bc = bn; a0c = a0n; a1c = a1n;
            }
            // ---- in-register LSTM cell for (all 32 rows) x (4 hidden units) ----
#pragma unroll
            for (int s = 0; s < 2; ++s) {
#pragma unroll
                for (int r = 0; r < 8; ++r) {
                    float v = accs[s][r] + biasv;
                    int base = lane & ~3;
                    float gi = __shfl(v, base + 0, 32);
                    float gf = __shfl(v, base + 1, 32);
                    float gg = __shfl(v, base + 2, 32);
                    float go = __shfl(v, base + 3, 32);
                    if (storer) {
                        int m = r + half * 8 + s * 16;     // batch row
                        float co = cbuf[(long)m * H_ + jGate];
                        float si = 1.0f / (1.0f + __expf(-gi));
                        float sf = 1.0f / (1.0f + __expf(-gf));
                        float so = 1.0f / (1.0f + __expf(-go));
                        float cn = sf * co + si * tanhf(gg);
                        float hn = so * tanhf(cn);
                        cbuf[(long)m * H_ + jGate] = cn;
                        xhn[(size_t)m * K4 + H_ + jGate] = (__bf16)hn;
                        if (t + 1 < T_)
                            xhn[(size_t)m * K4 + jGate] =
                                Xpre[((size_t)(t + 1) * B_ + m) * H_ + jGate];
                        if (t == T_ - 1) {
                            out[(size_t)B_ * T_ * S_ + (long)m * H_ + jGate]            = hn;
                            out[(size_t)B_ * T_ * S_ + B_ * H_ + (long)m * H_ + jGate]  = cn;
                        }
                    }
                }
            }
        }

        if (isDec && t >= 1) {
            // ---- decoder GEMM for step t-1 from resident h in xh[cur] ----
            const __bf16* Bp = Bdec + (size_t)nt * (H_ / 32) * 512 + lane * 16;
            const __bf16* arow0 = xh + H_ + (size_t)l16 * K4 + half * 8;
            const __bf16* arow1 = arow0 + (size_t)16 * K4;
            v8f accs[2] = {v8f{}, v8f{}};
            v16bf bc  = *(const v16bf*)Bp;
            v16bf a0c = load_afrag(arow0);
            v16bf a1c = load_afrag(arow1);
            for (int kt = 0; kt < H_ / 32; ++kt) {
                v16bf bn = bc, a0n = a0c, a1n = a1c;
                if (kt + 1 < H_ / 32) {
                    bn  = *(const v16bf*)(Bp + (size_t)(kt + 1) * 512);
                    a0n = load_afrag(arow0 + (kt + 1) * 32);
                    a1n = load_afrag(arow1 + (kt + 1) * 32);
                }
                accs[0] = __builtin_amdgcn_wmma_f32_16x16x32_bf16(false, a0c, false, bc, (short)0, accs[0], false, false);
                accs[1] = __builtin_amdgcn_wmma_f32_16x16x32_bf16(false, a1c, false, bc, (short)0, accs[1], false, false);
                bc = bn; a0c = a0n; a1c = a1n;
            }
            int s_col = nt * 16 + l16;
#pragma unroll
            for (int r = 0; r < 8; ++r) {
                int m = r + half * 8;
                out[((size_t)m        * T_ + (t - 1)) * S_ + s_col] = accs[0][r] + dbias;
                out[((size_t)(m + 16) * T_ + (t - 1)) * S_ + s_col] = accs[1][r] + dbias;
            }
        }

        if (t < T_) grid_sync(cnt, sense, (unsigned)(t + 1), nblk);
    }
}

// ---------------------------------------------------------------------------
extern "C" void kernel_launch(void* const* d_in, const int* in_sizes, int n_in,
                              void* d_out, int out_size, void* d_ws, size_t ws_size,
                              hipStream_t stream)
{
    (void)in_sizes; (void)n_in; (void)out_size; (void)ws_size;

    const float* input_1  = (const float*)d_in[0];
    const float* input_2  = (const float*)d_in[1];
    const float* h0       = (const float*)d_in[2];
    const float* c0       = (const float*)d_in[3];
    const int*   routers  = (const int*)  d_in[4];
    const float* m1_W     = (const float*)d_in[5];
    const float* m1_b     = (const float*)d_in[6];
    const float* m2_W     = (const float*)d_in[7];
    const float* m2_b     = (const float*)d_in[8];
    const float* enc_W    = (const float*)d_in[9];
    const float* enc_b    = (const float*)d_in[10];
    const float* lstm_Wih = (const float*)d_in[11];
    const float* lstm_Whh = (const float*)d_in[12];
    const float* lstm_bih = (const float*)d_in[13];
    const float* lstm_bhh = (const float*)d_in[14];
    const float* dec_W    = (const float*)d_in[15];
    const float* dec_b    = (const float*)d_in[16];

    float* out = (float*)d_out;

    // Workspace carve-up (all chunks 128B-aligned by construction)
    char* ws = (char*)d_ws;
    size_t ofs = 0;
    __bf16* Blstm = (__bf16*)(ws + ofs); ofs += (size_t)K4 * G4 * 2;          // 67.1 MB
    __bf16* Bdec  = (__bf16*)(ws + ofs); ofs += (size_t)H_ * S_ * 2;          //  1.0 MB
    __bf16* Xpre  = (__bf16*)(ws + ofs); ofs += (size_t)T_ * B_ * H_ * 2;     // 26.2 MB
    __bf16* xhbuf = (__bf16*)(ws + ofs); ofs += (size_t)2 * B_ * K4 * 2;      //  0.5 MB
    float*  cbuf  = (float*)(ws + ofs);  ofs += (size_t)B_ * H_ * 4;          //  0.26 MB
    float*  pbias = (float*)(ws + ofs);  ofs += (size_t)G4 * 4;               //  32 KB
    unsigned* cnt   = (unsigned*)(ws + ofs); ofs += 128;
    unsigned* sense = (unsigned*)(ws + ofs); ofs += 128;

    // One-time: pack weights (lstm gets the i/f/g/o column interleave).
    {
        long total = (long)K4 * G4;
        pack_b_kernel<<<(int)(total / 256), 256, 0, stream>>>(lstm_Wih, lstm_Whh,
                                                              H_, K4, G4, 1, Blstm);
    }
    {
        long total = (long)H_ * S_;
        pack_b_kernel<<<(int)(total / 256), 256, 0, stream>>>(dec_W, dec_W,
                                                              H_, H_, S_, 0, Bdec);
    }
    pbias_kernel<<<G4 / 256, 256, 0, stream>>>(lstm_bih, lstm_bhh, pbias);

    // One-time: all encoder outputs (recurrence-independent).
    encoder_pre_kernel<<<T_ * B_, 256, 0, stream>>>(input_1, input_2, routers,
                                                    m1_W, m1_b, m2_W, m2_b,
                                                    enc_W, enc_b, Xpre);

    init_kernel<<<(B_ * H_) / 256, 256, 0, stream>>>(h0, c0, Xpre, xhbuf, cbuf, cnt, sense);

    // Entire 200-step recurrence in one persistent launch:
    // 512 gate waves + 16 decoder waves = 528 waves = 66 blocks of 8 waves.
    dkt_persistent_kernel<<<66, 256, 0, stream>>>(Blstm, Bdec, pbias, dec_b, Xpre,
                                                  xhbuf, cbuf, out, cnt, sense);
}
